// DirectionalRecursiveBlock_74801150427531
// MI455X (gfx1250) — matrix operand order repaired and checked by reading, not compiled, not run on hardware
//
#include <hip/hip_runtime.h>
#include <hip/hip_bf16.h>
#include <math.h>

typedef __bf16 bf16;
typedef __attribute__((ext_vector_type(16))) __bf16 v16bf;
typedef __attribute__((ext_vector_type(8)))  __bf16 v8bf;
typedef __attribute__((ext_vector_type(8)))  float   v8f;
typedef __attribute__((ext_vector_type(4)))  unsigned int v4u;
typedef __attribute__((ext_vector_type(4)))  int v4i;
typedef __attribute__((ext_vector_type(8)))  int v8i;

#define DEV __device__ __forceinline__

#if __has_builtin(__builtin_amdgcn_tensor_load_to_lds)
#define USE_TDM 1
#else
#define USE_TDM 0
#endif

#if defined(__has_include)
#if __has_include(<hip/amd_detail/amd_gfx1250_TDM.h>)
#define TDM_SIX_ARGS 1
#else
#define TDM_SIX_ARGS 0
#endif
#else
#define TDM_SIX_ARGS 0
#endif

DEV v16bf make_v16(v8bf lo, v8bf hi) {
  return __builtin_shufflevector(lo, hi, 0,1,2,3,4,5,6,7,8,9,10,11,12,13,14,15);
}

// A-fragment (16x32 bf16, MxK): lane m = lane&15, holds K = kh..kh+7 and 16+kh..16+kh+7
// where kh = (lane>>4)*8.  Two contiguous 8-elem (16B) loads per lane.
DEV v16bf load_frag_a(const bf16* base, int ld, int k0, int lane) {
  int m  = lane & 15;
  int kh = (lane >> 4) * 8;
  const bf16* p = base + (size_t)m * ld + k0 + kh;
  v8bf lo = *(const v8bf*)p;
  v8bf hi = *(const v8bf*)(p + 16);
  return make_v16(lo, hi);
}

// B-fragment (32x16 bf16, KxN) from row-major W[n][k]: lane n = lane&15,
// holds K = kh..kh+7 and 16+kh..16+kh+7 where kh = (lane>>4)*8.  Two 16B loads.
DEV v16bf load_frag_b(const bf16* base, int ld, int k0, int lane) {
  int n  = lane & 15;
  int kh = (lane >> 4) * 8;
  const bf16* p = base + (size_t)n * ld + k0 + kh;
  return make_v16(*(const v8bf*)p, *(const v8bf*)(p + 16));
}

DEV v8f wmma_bf16(v16bf a, v16bf b, v8f c) {
  return __builtin_amdgcn_wmma_f32_16x16x32_bf16(
      /*neg_a=*/false, a, /*neg_b=*/false, b,
      /*c_mod=*/(short)0, c, /*reuse_a=*/false, /*reuse_b=*/false);
}

DEV v8f vzero8() {
  v8f z;
#pragma unroll
  for (int i = 0; i < 8; ++i) z[i] = 0.f;
  return z;
}

DEV v16bf ones16bf() {
  v16bf o;
#pragma unroll
  for (int i = 0; i < 16; ++i) o[i] = (bf16)1.0f;
  return o;
}

// ---- row-of-16 max reduction (XOR butterfly). Prefer v_permlane16_b32 (pure
// VALU, no LDS round trip) over ds_bpermute-based __shfl_xor.
#if __has_builtin(__builtin_amdgcn_permlane16)
DEV float rowmax_step(float v, unsigned s0, unsigned s1) {
  unsigned u = __builtin_amdgcn_permlane16(
      __float_as_uint(v), __float_as_uint(v), s0, s1, false, false);
  return fmaxf(v, __uint_as_float(u));
}
DEV float rowmax16(float v) {
  v = rowmax_step(v, 0x67452301u, 0xEFCDAB89u);  // xor 1
  v = rowmax_step(v, 0x54761032u, 0xDCFE98BAu);  // xor 2
  v = rowmax_step(v, 0x32107654u, 0xBA98FEDCu);  // xor 4
  v = rowmax_step(v, 0xFEDCBA98u, 0x76543210u);  // xor 8
  return v;
}
#else
DEV float rowmax16(float v) {
  for (int msk = 1; msk < 16; msk <<= 1)
    v = fmaxf(v, __shfl_xor(v, msk, 32));
  return v;
}
#endif

// ---------------------------------------------------------------------------
// Tensor Data Mover: DMA one 16x32-element bf16 tile (row stride K elements)
// from global memory into LDS.  D# built per CDNA5 ISA ch.8.
// ---------------------------------------------------------------------------
#if USE_TDM
DEV void tdm_stage_rows(const bf16* gsrc, const bf16* ldst, int K) {
  unsigned long long ga = (unsigned long long)(const void*)gsrc;
  unsigned lo = (unsigned)(size_t)(const void*)ldst;  // flat LDS addr[31:0] = LDS byte offset
  v4u g0;
  g0[0] = 1u;                                               // count=1, user descriptor
  g0[1] = lo;                                               // lds_addr
  g0[2] = (unsigned)ga;                                     // global_addr[31:0]
  g0[3] = (unsigned)((ga >> 32) & 0x01ffffffu) | 0x80000000u;  // addr[56:32] | type=2
  unsigned td0 = (unsigned)K;   // tensor dim0 (elements)
  unsigned td1 = 16u;           // tensor dim1 (rows)
  v8i g1;
  g1[0] = (int)(1u << 16);                                   // data_size = 1 (2 bytes)
  g1[1] = (int)((td0 & 0xffffu) << 16);                      // tensor_dim0[15:0]
  g1[2] = (int)(((td0 >> 16) & 0xffffu) | ((td1 & 0xffffu) << 16));
  g1[3] = (int)(((td1 >> 16) & 0xffffu) | (32u << 16));      // tile_dim0 = 32
  g1[4] = 16;                                                // tile_dim1=16, tile_dim2=0
  g1[5] = (int)td0;                                          // tensor_dim0_stride[31:0] = K
  g1[6] = 0;
  g1[7] = 0;
  v4i z4 = {0, 0, 0, 0};
#if TDM_SIX_ARGS
  v8i z8 = {0, 0, 0, 0, 0, 0, 0, 0};
  __builtin_amdgcn_tensor_load_to_lds(g0, g1, z4, z4, z8, 0);
#else
  __builtin_amdgcn_tensor_load_to_lds(g0, g1, z4, z4, 0);
#endif
}
#endif

DEV void tdm_wait0() {
#if __has_builtin(__builtin_amdgcn_s_wait_tensorcnt)
  __builtin_amdgcn_s_wait_tensorcnt((short)0);
#else
  asm volatile("s_wait_tensorcnt 0" ::: "memory");
#endif
}

// Stage the block's 128x32 bf16 A-panel into LDS (wave wv owns 16 rows).
DEV void stage_a_tile(const bf16* __restrict__ A, int m0blk, int K, int k0,
                      bf16* ldsbuf, int wv, int tid) {
#if USE_TDM
  (void)tid;
  tdm_stage_rows(A + (size_t)(m0blk + wv * 16) * K + k0, ldsbuf + wv * 16 * 32, K);
#else
  for (int e = tid; e < 128 * 16; e += 256) {
    int row = e >> 4, c = e & 15;
    *(unsigned*)&ldsbuf[row * 32 + c * 2] =
        *(const unsigned*)(A + (size_t)(m0blk + row) * K + k0 + c * 2);
  }
#endif
}

// ---------------------------------------------------------------------------
// float -> bf16 elementwise conversion
// ---------------------------------------------------------------------------
__global__ __launch_bounds__(256) void cvt_bf16(const float* __restrict__ in,
                                                bf16* __restrict__ out, size_t n) {
  size_t i = (size_t)blockIdx.x * 256 + threadIdx.x;
  if (i < n) out[i] = (bf16)in[i];
}

// ---------------------------------------------------------------------------
// adaLN gamma/beta: gb[b][j] = emb[b] . w[j] + bias[j]   (2 x 2048 outputs)
// ---------------------------------------------------------------------------
__global__ __launch_bounds__(256) void adaln_gb(const float* __restrict__ emb,
                                                const float* __restrict__ w,
                                                const float* __restrict__ bias,
                                                float* __restrict__ gb) {
  int t = blockIdx.x * 256 + threadIdx.x;  // 0 .. 2*2048-1
  int b = t >> 11, j = t & 2047;
  const float* e  = emb + (size_t)b * 1024;
  const float* wr = w + (size_t)j * 1024;
  float s = 0.f;
  for (int d = 0; d < 1024; ++d) s += e[d] * wr[d];
  gb[t] = s + bias[j];
}

// ---------------------------------------------------------------------------
// LayerNorm + adaLN modulation, f32 in -> bf16 out. One block per row.
// ---------------------------------------------------------------------------
__global__ __launch_bounds__(256) void ln_mod(const float* __restrict__ x,
                                              const float* __restrict__ gb,
                                              bf16* __restrict__ out, int L) {
  int row = blockIdx.x;
  int b   = row / L;
  const float* xr = x + (size_t)row * 1024;
  __shared__ float s1[256], s2[256];
  int tid = threadIdx.x;
  float a = 0.f, q = 0.f;
  for (int j = tid; j < 1024; j += 256) {
    float v = xr[j];
    a += v; q += v * v;
  }
  s1[tid] = a; s2[tid] = q;
  __syncthreads();
  for (int st = 128; st > 0; st >>= 1) {
    if (tid < st) { s1[tid] += s1[tid + st]; s2[tid] += s2[tid + st]; }
    __syncthreads();
  }
  float mu  = s1[0] * (1.f / 1024.f);
  float var = s2[0] * (1.f / 1024.f) - mu * mu;
  float rs  = rsqrtf(var + 1e-5f);
  const float* g = gb + (size_t)b * 2048;
  bf16* o = out + (size_t)row * 1024;
  for (int j = tid; j < 1024; j += 256) {
    float v = (xr[j] - mu) * rs;
    o[j] = (bf16)(v * (1.f + g[j]) + g[1024 + j]);
  }
}

// ---------------------------------------------------------------------------
// RoPE + reshape to [B*H][Lx][64] bf16 (heads split out of a f32 GEMM output)
// ---------------------------------------------------------------------------
__global__ __launch_bounds__(256) void rope_reshape(const float* __restrict__ src,
                                                    int srcStride, int colOff,
                                                    bf16* __restrict__ dst,
                                                    int Lx, int applyRope) {
  size_t t = (size_t)blockIdx.x * 256 + threadIdx.x;  // B*H*Lx*32 threads
  int d       = (int)(t & 31);
  size_t rest = t >> 5;
  int l  = (int)(rest % Lx);
  int bh = (int)(rest / Lx);
  int b = bh >> 4, h = bh & 15;
  const float* p = src + ((size_t)b * Lx + l) * srcStride + colOff + h * 64;
  float x1 = p[d], x2 = p[d + 32];
  float o1 = x1, o2 = x2;
  if (applyRope) {
    float freq = powf(10000.f, -(float)d * (1.f / 32.f));
    float ang  = (float)l * freq;
    float s = sinf(ang), c = cosf(ang);
    o1 = x1 * c - x2 * s;
    o2 = x2 * c + x1 * s;
  }
  bf16* q = dst + ((size_t)bh * Lx + l) * 64;
  q[d] = (bf16)o1;
  q[d + 32] = (bf16)o2;
}

// ---------------------------------------------------------------------------
// V transpose: f32 [B][Lx][D] (head cols) -> bf16 Vt [B*H][64][Lx]
// ---------------------------------------------------------------------------
__global__ __launch_bounds__(256) void vtrans(const float* __restrict__ src,
                                              int srcStride, int colOff,
                                              bf16* __restrict__ dst, int Lx) {
  size_t t = (size_t)blockIdx.x * 256 + threadIdx.x;  // B*H*Lx*64 threads
  int d       = (int)(t & 63);
  size_t rest = t >> 6;
  int l  = (int)(rest % Lx);
  int bh = (int)(rest / Lx);
  int b = bh >> 4, h = bh & 15;
  float v = src[((size_t)b * Lx + l) * srcStride + colOff + h * 64 + d];
  dst[((size_t)bh * 64 + d) * Lx + l] = (bf16)v;
}

// ---------------------------------------------------------------------------
// bf16 WMMA GEMM: C[M][N] = A[M][K] * W[N][K]^T (+ epilogue), f32 accumulate.
// Wave = 64x64 tile (4x4 of 16x16 frags). Block = 8 waves = 128x256 tile.
// A-panel (128x32 per K step) staged into LDS by the Tensor Data Mover,
// double-buffered so the DMA of tile k+1 overlaps WMMAs on tile k.
// EPI: 0 = store f32; 1 = +res f32; 2 = gelu(C+bias) -> bf16; 3 = C+bias+res f32
// ---------------------------------------------------------------------------
template <int EPI>
__global__ __launch_bounds__(256) void gemm_xwT(const bf16* __restrict__ A,
                                                const bf16* __restrict__ W,
                                                int M, int N, int K,
                                                float* __restrict__ outf,
                                                bf16* __restrict__ outb,
                                                const float* __restrict__ res,
                                                const float* __restrict__ bias) {
  __shared__ bf16 ldsA[2][128 * 32];  // 2 x 8KB double buffer
  const int tid  = threadIdx.x;
  const int lane = tid & 31;
  const int wv   = tid >> 5;
  const int m0blk = blockIdx.y * 128;
  const int m0   = m0blk + (wv >> 2) * 64;
  const int n0   = blockIdx.x * 256 + (wv & 3) * 64;
  const int half = lane >> 4, nl = lane & 15;

  v8f acc[4][4];
#pragma unroll
  for (int i = 0; i < 4; ++i)
#pragma unroll
    for (int j = 0; j < 4; ++j) acc[i][j] = vzero8();

  // prologue: stage first A tile
  stage_a_tile(A, m0blk, K, 0, &ldsA[0][0], wv, tid);
#if USE_TDM
  tdm_wait0();
#endif
  __syncthreads();

  for (int k0 = 0; k0 < K; k0 += 32) {
    const int  cur  = (k0 >> 5) & 1;
    const bool more = (k0 + 32) < K;
    if (more)  // async DMA of next tile overlaps this tile's math
      stage_a_tile(A, m0blk, K, k0 + 32, &ldsA[cur ^ 1][0], wv, tid);
    if (k0 + 128 < K)  // L2 prefetch for the streamed W operand
      __builtin_prefetch(W + (size_t)(n0 + nl) * K + k0 + 128, 0, 1);

    v16bf af[4], bfr[4];
#pragma unroll
    for (int i = 0; i < 4; ++i)
      af[i] = load_frag_a(&ldsA[cur][((wv >> 2) * 64 + 16 * i) * 32], 32, 0, lane);
#pragma unroll
    for (int j = 0; j < 4; ++j)
      bfr[j] = load_frag_b(W + (size_t)(n0 + 16 * j) * K, K, k0, lane);
#pragma unroll
    for (int i = 0; i < 4; ++i)
#pragma unroll
      for (int j = 0; j < 4; ++j)
        acc[i][j] = wmma_bf16(af[i], bfr[j], acc[i][j]);

    if (more) {
#if USE_TDM
      tdm_wait0();
#endif
      __syncthreads();
    }
  }

#pragma unroll
  for (int i = 0; i < 4; ++i) {
#pragma unroll
    for (int r = 0; r < 8; ++r) {
      size_t row = (size_t)(m0 + 16 * i + r + 8 * half);
#pragma unroll
      for (int j = 0; j < 4; ++j) {
        int col    = n0 + 16 * j + nl;
        size_t idx = row * (size_t)N + col;
        float v    = acc[i][j][r];
        if constexpr (EPI == 0) {
          outf[idx] = v;
        } else if constexpr (EPI == 1) {
          outf[idx] = v + res[idx];
        } else if constexpr (EPI == 2) {
          v += bias[col];
          outb[idx] = (bf16)(0.5f * v * (1.f + erff(v * 0.70710678118f)));
        } else {
          outf[idx] = v + bias[col] + res[idx];
        }
      }
    }
  }
}

// ---------------------------------------------------------------------------
// Flash attention (wave32 WMMA). One wave = 32 query rows of one (b,h).
// Q: [B*H][Lq][64] bf16, K: [B*H][Lk][64] bf16, Vt: [B*H][64][Lk] bf16.
// Out: bf16 [B*Lq][1024] with col = h*64+dh (ready to be the next GEMM's A).
// Row-sum l is accumulated as an extra WMMA against a ones B-matrix (the
// correction rescale applied to O applies identically to l), so only the
// row-max needs a cross-lane reduction (v_permlane16_b32 butterflies).
// ---------------------------------------------------------------------------
__global__ __launch_bounds__(256) void flash_attn(const bf16* __restrict__ Q,
                                                  const bf16* __restrict__ Kb,
                                                  const bf16* __restrict__ Vt,
                                                  bf16* __restrict__ Out,
                                                  int Lq, int Lk, float scale) {
  __shared__ bf16 ldsP[8 * 32 * 32];  // 2KB per wave, private P tile
  const int lane = threadIdx.x & 31;
  const int wv   = threadIdx.x >> 5;
  const int gw   = blockIdx.x * 8 + wv;
  const int qtiles = Lq >> 5;
  const int bh = gw / qtiles;
  const int qt = gw % qtiles;
  const int b = bh >> 4, h = bh & 15;
  const int half = lane >> 4, nl = lane & 15;

  const bf16* qp = Q + ((size_t)bh * Lq + qt * 32) * 64;
  const bf16* kp = Kb + (size_t)bh * Lk * 64;
  const bf16* vp = Vt + (size_t)bh * 64 * Lk;
  bf16* myP = &ldsP[wv * 32 * 32];
  const v16bf vones = ones16bf();

  v16bf QA[2][2];
#pragma unroll
  for (int mf = 0; mf < 2; ++mf)
#pragma unroll
    for (int ks = 0; ks < 2; ++ks)
      QA[mf][ks] = load_frag_a(qp + mf * 16 * 64, 64, ks * 32, lane);

  v8f Oa[2][4], La[2];
  float mrow[2][8];
#pragma unroll
  for (int mf = 0; mf < 2; ++mf) {
#pragma unroll
    for (int nf = 0; nf < 4; ++nf) Oa[mf][nf] = vzero8();
    La[mf] = vzero8();
#pragma unroll
    for (int r = 0; r < 8; ++r) mrow[mf][r] = -1e30f;
  }

  for (int t0 = 0; t0 < Lk; t0 += 32) {
    // ---- S = Q * K^T  (64 dh contraction = 2 wmma k-steps) ----
    v8f S[2][2];
#pragma unroll
    for (int mf = 0; mf < 2; ++mf)
#pragma unroll
      for (int tf = 0; tf < 2; ++tf) S[mf][tf] = vzero8();
#pragma unroll
    for (int ks = 0; ks < 2; ++ks) {
      v16bf KB0[2];
#pragma unroll
      for (int tf = 0; tf < 2; ++tf)
        KB0[tf] = load_frag_b(kp + (size_t)(t0 + tf * 16) * 64, 64, ks * 32, lane);
#pragma unroll
      for (int mf = 0; mf < 2; ++mf)
#pragma unroll
        for (int tf = 0; tf < 2; ++tf)
          S[mf][tf] = wmma_bf16(QA[mf][ks], KB0[tf], S[mf][tf]);
    }

    // ---- online softmax: only row-max needs a cross-lane reduction ----
#pragma unroll
    for (int mf = 0; mf < 2; ++mf) {
      float rm[8];
#pragma unroll
      for (int r = 0; r < 8; ++r) {
        float a = S[mf][0][r] * scale, c = S[mf][1][r] * scale;
        S[mf][0][r] = a; S[mf][1][r] = c;
        rm[r] = rowmax16(fmaxf(a, c));
      }
#pragma unroll
      for (int r = 0; r < 8; ++r) {
        float mn   = fmaxf(mrow[mf][r], rm[r]);
        float corr = expf(mrow[mf][r] - mn);
        mrow[mf][r] = mn;
        S[mf][0][r] = expf(S[mf][0][r] - mn);
        S[mf][1][r] = expf(S[mf][1][r] - mn);
        La[mf][r] *= corr;
#pragma unroll
        for (int nf = 0; nf < 4; ++nf) Oa[mf][nf][r] *= corr;
      }
      // spill P (C-layout) to this wave's LDS tile as bf16 [32 rows][32 cols]
#pragma unroll
      for (int tf = 0; tf < 2; ++tf)
#pragma unroll
        for (int r = 0; r < 8; ++r)
          myP[(mf * 16 + r + 8 * half) * 32 + tf * 16 + nl] = (bf16)S[mf][tf][r];
    }

    __builtin_amdgcn_wave_barrier();
    asm volatile("s_wait_dscnt 0" ::: "memory");  // CDNA5 split LDS counter
    __builtin_amdgcn_wave_barrier();

    // ---- re-fragment P as A-matrix; O += P*V, L += P*ones (K = 32 tokens) --
    v16bf Pa[2];
#pragma unroll
    for (int mf = 0; mf < 2; ++mf) {
      const bf16* pp = myP + (mf * 16 + nl) * 32 + half * 8;
      Pa[mf] = make_v16(*(const v8bf*)pp, *(const v8bf*)(pp + 16));
      La[mf] = wmma_bf16(Pa[mf], vones, La[mf]);  // running row-sum
    }
#pragma unroll
    for (int nf = 0; nf < 4; ++nf) {
      const bf16* vq = vp + (size_t)(nf * 16 + nl) * Lk + t0 + half * 8;
      v16bf VB = make_v16(*(const v8bf*)vq, *(const v8bf*)(vq + 16));
#pragma unroll
      for (int mf = 0; mf < 2; ++mf)
        Oa[mf][nf] = wmma_bf16(Pa[mf], VB, Oa[mf][nf]);
    }
  }

  // ---- epilogue: O /= l, store bf16 at [b*Lq+q][h*64+dh] ----
#pragma unroll
  for (int mf = 0; mf < 2; ++mf)
#pragma unroll
    for (int r = 0; r < 8; ++r) {
      float inv  = 1.f / La[mf][r];
      size_t row = (size_t)b * Lq + qt * 32 + mf * 16 + r + 8 * half;
      bf16* orow = Out + row * 1024 + h * 64;
#pragma unroll
      for (int nf = 0; nf < 4; ++nf)
        orow[nf * 16 + nl] = (bf16)(Oa[mf][nf][r] * inv);
    }
}

// ---------------------------------------------------------------------------
// Host launcher
// ---------------------------------------------------------------------------
extern "C" void kernel_launch(void* const* d_in, const int* in_sizes, int n_in,
                              void* d_out, int out_size, void* d_ws, size_t ws_size,
                              hipStream_t stream) {
  (void)in_sizes; (void)n_in; (void)out_size; (void)ws_size;
  const float* x       = (const float*)d_in[0];
  const float* emb     = (const float*)d_in[1];
  const float* ctx     = (const float*)d_in[2];
  const float* ln1_w   = (const float*)d_in[3];
  const float* ln1_b   = (const float*)d_in[4];
  const float* ln2_w   = (const float*)d_in[5];
  const float* ln2_b   = (const float*)d_in[6];
  const float* ln3_w   = (const float*)d_in[7];
  const float* ln3_b   = (const float*)d_in[8];
  const float* qkv_w   = (const float*)d_in[9];
  const float* sa_out_w= (const float*)d_in[10];
  const float* q_w     = (const float*)d_in[11];
  const float* k_w     = (const float*)d_in[12];
  const float* v_w     = (const float*)d_in[13];
  const float* ca_out_w= (const float*)d_in[14];
  const float* ffn_w1  = (const float*)d_in[15];
  const float* ffn_b1  = (const float*)d_in[16];
  const float* ffn_w2  = (const float*)d_in[17];
  const float* ffn_b2  = (const float*)d_in[18];
  float* out = (float*)d_out;

  constexpr int B = 2, L = 2048, Lc = 1024, D = 1024, H = 16, DFF = 4096;
  constexpr int M = B * L;     // 4096
  constexpr int Mc = B * Lc;   // 2048

  uint8_t* base = (uint8_t*)d_ws;
  size_t off = 0;
  auto alloc = [&](size_t bytes) -> void* {
    void* p = base + off;
    off = (off + bytes + 255) & ~(size_t)255;
    return p;
  };
  bf16* wb_qkv = (bf16*)alloc((size_t)3 * D * D * 2);
  bf16* wb_sa  = (bf16*)alloc((size_t)D * D * 2);
  bf16* wb_q   = (bf16*)alloc((size_t)D * D * 2);
  bf16* wb_k   = (bf16*)alloc((size_t)D * D * 2);
  bf16* wb_v   = (bf16*)alloc((size_t)D * D * 2);
  bf16* wb_ca  = (bf16*)alloc((size_t)D * D * 2);
  bf16* wb_f1  = (bf16*)alloc((size_t)DFF * D * 2);
  bf16* wb_f2  = (bf16*)alloc((size_t)D * DFF * 2);
  bf16* ctx_bf = (bf16*)alloc((size_t)Mc * D * 2);
  bf16* a_bf   = (bf16*)alloc((size_t)M * D * 2);
  float* gb1   = (float*)alloc((size_t)B * 2 * D * 4);
  float* gb2   = (float*)alloc((size_t)B * 2 * D * 4);
  float* gb3   = (float*)alloc((size_t)B * 2 * D * 4);
  float* s1    = (float*)alloc((size_t)M * 3 * D * 4);   // biggest f32 scratch
  bf16* qh     = (bf16*)alloc((size_t)M * D * 2);        // [B*H][Lq][64]
  bf16* kh     = (bf16*)alloc((size_t)M * D * 2);        // [B*H][Lk][64]
  bf16* vT     = (bf16*)alloc((size_t)M * D * 2);        // [B*H][64][Lk]
  bf16* ao     = (bf16*)alloc((size_t)M * D * 2);        // attn out bf16
  float* X     = (float*)alloc((size_t)M * D * 4);       // running residual
  bf16* hbf    = (bf16*)alloc((size_t)M * DFF * 2);      // gelu hidden

  auto cvt = [&](const float* in, bf16* o, size_t n) {
    cvt_bf16<<<dim3((unsigned)((n + 255) / 256)), dim3(256), 0, stream>>>(in, o, n);
  };

  // -- weight / context conversion to bf16 (deterministic each call) --
  cvt(qkv_w,    wb_qkv, (size_t)3 * D * D);
  cvt(sa_out_w, wb_sa,  (size_t)D * D);
  cvt(q_w,      wb_q,   (size_t)D * D);
  cvt(k_w,      wb_k,   (size_t)D * D);
  cvt(v_w,      wb_v,   (size_t)D * D);
  cvt(ca_out_w, wb_ca,  (size_t)D * D);
  cvt(ffn_w1,   wb_f1,  (size_t)DFF * D);
  cvt(ffn_w2,   wb_f2,  (size_t)D * DFF);
  cvt(ctx,      ctx_bf, (size_t)Mc * D);

  const float scale = 0.125f;  // 1/sqrt(64)

  // ================= Stage 1: self-attention =================
  adaln_gb<<<16, 256, 0, stream>>>(emb, ln1_w, ln1_b, gb1);
  ln_mod<<<M, 256, 0, stream>>>(x, gb1, a_bf, L);
  gemm_xwT<0><<<dim3(3 * D / 256, M / 128), 256, 0, stream>>>(
      a_bf, wb_qkv, M, 3 * D, D, s1, nullptr, nullptr, nullptr);
  rope_reshape<<<4 * L, 256, 0, stream>>>(s1, 3 * D, 0,     qh, L, 1);
  rope_reshape<<<4 * L, 256, 0, stream>>>(s1, 3 * D, D,     kh, L, 1);
  vtrans<<<8 * L, 256, 0, stream>>>(s1, 3 * D, 2 * D, vT, L);
  flash_attn<<<(B * H * (L / 32)) / 8, 256, 0, stream>>>(qh, kh, vT, ao, L, L, scale);
  gemm_xwT<1><<<dim3(D / 256, M / 128), 256, 0, stream>>>(
      ao, wb_sa, M, D, D, X, nullptr, x, nullptr);

  // ================= Stage 2: cross-attention =================
  adaln_gb<<<16, 256, 0, stream>>>(emb, ln2_w, ln2_b, gb2);
  ln_mod<<<M, 256, 0, stream>>>(X, gb2, a_bf, L);
  float* s1q = s1;
  float* s1k = s1 + (size_t)M * D;
  float* s1v = s1k + (size_t)Mc * D;
  gemm_xwT<0><<<dim3(D / 256, M / 128), 256, 0, stream>>>(
      a_bf, wb_q, M, D, D, s1q, nullptr, nullptr, nullptr);
  gemm_xwT<0><<<dim3(D / 256, Mc / 128), 256, 0, stream>>>(
      ctx_bf, wb_k, Mc, D, D, s1k, nullptr, nullptr, nullptr);
  gemm_xwT<0><<<dim3(D / 256, Mc / 128), 256, 0, stream>>>(
      ctx_bf, wb_v, Mc, D, D, s1v, nullptr, nullptr, nullptr);
  rope_reshape<<<4 * L,  256, 0, stream>>>(s1q, D, 0, qh, L,  1);
  rope_reshape<<<4 * Lc, 256, 0, stream>>>(s1k, D, 0, kh, Lc, 1);
  vtrans<<<8 * Lc, 256, 0, stream>>>(s1v, D, 0, vT, Lc);
  flash_attn<<<(B * H * (L / 32)) / 8, 256, 0, stream>>>(qh, kh, vT, ao, L, Lc, scale);
  gemm_xwT<1><<<dim3(D / 256, M / 128), 256, 0, stream>>>(
      ao, wb_ca, M, D, D, X, nullptr, X, nullptr);

  // ================= Stage 3: FFN =================
  adaln_gb<<<16, 256, 0, stream>>>(emb, ln3_w, ln3_b, gb3);
  ln_mod<<<M, 256, 0, stream>>>(X, gb3, a_bf, L);
  gemm_xwT<2><<<dim3(DFF / 256, M / 128), 256, 0, stream>>>(
      a_bf, wb_f1, M, DFF, D, nullptr, hbf, nullptr, ffn_b1);
  gemm_xwT<3><<<dim3(D / 256, M / 128), 256, 0, stream>>>(
      hbf, wb_f2, M, D, DFF, out, nullptr, X, ffn_b2);
}
